// MultiScaleLinearAttention_11098195493345
// MI455X (gfx1250) — matrix operand order repaired and checked by reading, not compiled, not run on hardware
//
#include <hip/hip_runtime.h>
#include <hip/hip_bf16.h>

typedef __attribute__((ext_vector_type(16))) __bf16 v16bf;
typedef __attribute__((ext_vector_type(8)))  __bf16 v8bf;
typedef __attribute__((ext_vector_type(2)))  __bf16 v2bf;
typedef __attribute__((ext_vector_type(8)))  float  v8f;

#define NTOT   16384   // H*W
#define HH     128
#define WW     128
#define CC     256
#define NHEADS 8
#define DHEAD  32
#define BB     4
#define BH     32                       // B * heads
#define FSCALE 0.17677669529663687f    // 32^-0.5

__device__ __forceinline__ v8f wmma_bf16(v16bf a, v16bf b, v8f c) {
  return __builtin_amdgcn_wmma_f32_16x16x32_bf16(false, a, false, b, (short)0, c,
                                                 false, false);
}

// elu(x*s) + 1  ==  x*s+1 (x>0)  else exp(x*s)
__device__ __forceinline__ float featmap(float x) {
  float y = x * FSCALE;
  return y > 0.f ? y + 1.f : __expf(y);
}

// ---------------------------------------------------------------- softmax(3)
__global__ void softmax3_kernel(const float* __restrict__ w, float* __restrict__ sw) {
  if (threadIdx.x == 0 && blockIdx.x == 0) {
    float m = fmaxf(w[0], fmaxf(w[1], w[2]));
    float e0 = __expf(w[0] - m), e1 = __expf(w[1] - m), e2 = __expf(w[2] - m);
    float s = e0 + e1 + e2;
    sw[0] = e0 / s; sw[1] = e1 / s; sw[2] = e2 / s;
  }
}

// --------------------------------------------------- channel GEMM (1x1 conv)
// Y[b][m][n] = sum_k Wm[m][k] * X[b][k][n],  K fixed = 256.
// Block = 256 threads (8 waves) -> 64(M) x 128(N) tile.
// Double-buffered LDS: stage tile k0+32 (coalesced float4 + packed v2bf
// ds_store_b32) while WMMAs consume tile k0. All 8 A fragments hoisted.
#define GK  256
#define BM  64
#define BN  128
#define LDS_STR 48   // padded row length in bf16 (32 data + 16 pad, 16B-aligned)

__global__ void __launch_bounds__(256)
gemm_wmma_kernel(const float* __restrict__ Wm, const float* __restrict__ X,
                 float* __restrict__ Y, int M, int Nn) {
  __shared__ __align__(16) __bf16 ldsX[2][BN * LDS_STR];   // 2 x 12 KB

  const int tid  = threadIdx.x;
  const int wv   = tid >> 5;
  const int lane = tid & 31;
  const int half = lane >> 4;
  const int r    = lane & 15;
  const int mw   = wv >> 1;        // 0..3 : 16-row slice within BM
  const int nw   = wv & 1;         // 0..1 : 64-col slice within BN

  const int mblocks = M / BM;
  const int nblocks = Nn / BN;
  int blk = blockIdx.x;
  int b   = blk / (mblocks * nblocks);
  int rem = blk - b * (mblocks * nblocks);
  int mb  = rem / nblocks;
  int nb  = rem - mb * nblocks;

  const float* Xb = X + (size_t)b * GK * Nn + (size_t)nb * BN;
  const float* Wr = Wm + (size_t)(mb * BM + mw * 16 + r) * GK;

  // stage X[k0..k0+32)[0..BN) -> ldsX[buf] transposed [n][k], bf16, k-pair packed
  auto stage = [&](int k0, int buf) {
#pragma unroll
    for (int it = 0; it < 2; ++it) {
      int g  = tid + it * 256;          // 512 groups = 16 k-pairs x 32 n-quads
      int kk = (g >> 5) << 1;           // even k: 0,2,...,30
      int n4 = (g & 31) << 2;           // 0,4,...,124
      float4 x0 = *(const float4*)(Xb + (size_t)(k0 + kk) * Nn + n4);
      float4 x1 = *(const float4*)(Xb + (size_t)(k0 + kk + 1) * Nn + n4);
      __bf16* base = &ldsX[buf][0];
      v2bf p;
      p[0] = (__bf16)x0.x; p[1] = (__bf16)x1.x;
      *(v2bf*)&base[(n4 + 0) * LDS_STR + kk] = p;
      p[0] = (__bf16)x0.y; p[1] = (__bf16)x1.y;
      *(v2bf*)&base[(n4 + 1) * LDS_STR + kk] = p;
      p[0] = (__bf16)x0.z; p[1] = (__bf16)x1.z;
      *(v2bf*)&base[(n4 + 2) * LDS_STR + kk] = p;
      p[0] = (__bf16)x0.w; p[1] = (__bf16)x1.w;
      *(v2bf*)&base[(n4 + 3) * LDS_STR + kk] = p;
    }
  };

  // ---- hoist all 8 A fragments (W rows contiguous, L2-resident) ----
  v16bf afr[GK / 32];
#pragma unroll
  for (int kt = 0; kt < GK / 32; ++kt) {
    const float* ar = Wr + kt * 32 + half * 8;
    float4 w0 = *(const float4*)(ar);
    float4 w1 = *(const float4*)(ar + 4);
    float4 w2 = *(const float4*)(ar + 16);
    float4 w3 = *(const float4*)(ar + 20);
    afr[kt][0]  = (__bf16)w0.x; afr[kt][1]  = (__bf16)w0.y;
    afr[kt][2]  = (__bf16)w0.z; afr[kt][3]  = (__bf16)w0.w;
    afr[kt][4]  = (__bf16)w1.x; afr[kt][5]  = (__bf16)w1.y;
    afr[kt][6]  = (__bf16)w1.z; afr[kt][7]  = (__bf16)w1.w;
    afr[kt][8]  = (__bf16)w2.x; afr[kt][9]  = (__bf16)w2.y;
    afr[kt][10] = (__bf16)w2.z; afr[kt][11] = (__bf16)w2.w;
    afr[kt][12] = (__bf16)w3.x; afr[kt][13] = (__bf16)w3.y;
    afr[kt][14] = (__bf16)w3.z; afr[kt][15] = (__bf16)w3.w;
  }

  v8f acc0 = {}, acc1 = {}, acc2 = {}, acc3 = {};

  stage(0, 0);
  __syncthreads();

#pragma unroll
  for (int kt = 0; kt < GK / 32; ++kt) {
    const int cur = kt & 1;
    if (kt + 1 < GK / 32) stage((kt + 1) * 32, cur ^ 1);   // overlap next tile

    // ---- all 4 B fragments from LDS, then 4 WMMAs back-to-back ----
    v16bf bfr[4];
#pragma unroll
    for (int t = 0; t < 4; ++t) {
      int n = nw * 64 + t * 16 + r;
      const v8bf* p = (const v8bf*)&ldsX[cur][n * LDS_STR + half * 8];
      v8bf x0 = p[0];
      v8bf x1 = p[2];
#pragma unroll
      for (int j = 0; j < 8; ++j) { bfr[t][j] = x0[j]; bfr[t][8 + j] = x1[j]; }
    }
    acc0 = wmma_bf16(afr[kt], bfr[0], acc0);
    acc1 = wmma_bf16(afr[kt], bfr[1], acc1);
    acc2 = wmma_bf16(afr[kt], bfr[2], acc2);
    acc3 = wmma_bf16(afr[kt], bfr[3], acc3);
    __syncthreads();   // one barrier per k-step (double buffered)
  }

  float* Yb = Y + (size_t)b * M * Nn
            + (size_t)(mb * BM + mw * 16 + half * 8) * Nn
            + (size_t)nb * BN + nw * 64 + r;
#pragma unroll
  for (int r8 = 0; r8 < 8; ++r8) {
    Yb[(size_t)r8 * Nn +  0] = acc0[r8];
    Yb[(size_t)r8 * Nn + 16] = acc1[r8];
    Yb[(size_t)r8 * Nn + 32] = acc2[r8];
    Yb[(size_t)r8 * Nn + 48] = acc3[r8];
  }
}

// --------------------------------------------------- kv = k'^T v , k_sum
// kv[bh][d][e] (32x32 f32) accumulated over N via WMMA; k_sum via a ones-B
// fragment. Per-lane fragment rows are contiguous -> float4 loads.
#define KV_SLICES 16
__global__ void __launch_bounds__(256)
kv_reduce_kernel(const float* __restrict__ qkv, float* __restrict__ kvb,
                 float* __restrict__ ksum) {
  const int wave = (int)((blockIdx.x * blockDim.x + threadIdx.x) >> 5);
  const int lane = threadIdx.x & 31;
  const int half = lane >> 4;
  const int r    = lane & 15;
  int bh    = wave / KV_SLICES;
  int slice = wave - bh * KV_SLICES;
  int b = bh >> 3, hh = bh & 7;
  const float* kbase = qkv + ((size_t)b * 3 * CC +     CC + hh * DHEAD) * NTOT;
  const float* vbase = qkv + ((size_t)b * 3 * CC + 2 * CC + hh * DHEAD) * NTOT;

  v16bf ones;
#pragma unroll
  for (int j = 0; j < 16; ++j) ones[j] = (__bf16)1.0f;

  v8f acc00 = {}, acc01 = {}, acc10 = {}, acc11 = {}, ks0 = {}, ks1 = {};
  const int nPer = NTOT / KV_SLICES;
  const int nEnd = (slice + 1) * nPer;
  for (int n0 = slice * nPer; n0 < nEnd; n0 += 32) {
    v16bf a0, a1;   // A = k'[d, n] tiles (d rows 0-15 / 16-31)
    {
      const float* kr = kbase + (size_t)r * NTOT + n0 + half * 8;
      float4 p0 = *(const float4*)(kr);
      float4 p1 = *(const float4*)(kr + 4);
      float4 p2 = *(const float4*)(kr + 16);
      float4 p3 = *(const float4*)(kr + 20);
      a0[0]=(__bf16)featmap(p0.x); a0[1]=(__bf16)featmap(p0.y); a0[2]=(__bf16)featmap(p0.z); a0[3]=(__bf16)featmap(p0.w);
      a0[4]=(__bf16)featmap(p1.x); a0[5]=(__bf16)featmap(p1.y); a0[6]=(__bf16)featmap(p1.z); a0[7]=(__bf16)featmap(p1.w);
      a0[8]=(__bf16)featmap(p2.x); a0[9]=(__bf16)featmap(p2.y); a0[10]=(__bf16)featmap(p2.z); a0[11]=(__bf16)featmap(p2.w);
      a0[12]=(__bf16)featmap(p3.x); a0[13]=(__bf16)featmap(p3.y); a0[14]=(__bf16)featmap(p3.z); a0[15]=(__bf16)featmap(p3.w);
      kr += (size_t)16 * NTOT;
      p0 = *(const float4*)(kr);
      p1 = *(const float4*)(kr + 4);
      p2 = *(const float4*)(kr + 16);
      p3 = *(const float4*)(kr + 20);
      a1[0]=(__bf16)featmap(p0.x); a1[1]=(__bf16)featmap(p0.y); a1[2]=(__bf16)featmap(p0.z); a1[3]=(__bf16)featmap(p0.w);
      a1[4]=(__bf16)featmap(p1.x); a1[5]=(__bf16)featmap(p1.y); a1[6]=(__bf16)featmap(p1.z); a1[7]=(__bf16)featmap(p1.w);
      a1[8]=(__bf16)featmap(p2.x); a1[9]=(__bf16)featmap(p2.y); a1[10]=(__bf16)featmap(p2.z); a1[11]=(__bf16)featmap(p2.w);
      a1[12]=(__bf16)featmap(p3.x); a1[13]=(__bf16)featmap(p3.y); a1[14]=(__bf16)featmap(p3.z); a1[15]=(__bf16)featmap(p3.w);
    }
    v16bf b0, b1;   // B = v[n, e]: lane = e column, elems = n 8*half+0..7, then 16+8*half+0..7
    {
      const float* vr = vbase + (size_t)r * NTOT + n0 + half * 8;
      float4 p0 = *(const float4*)(vr);
      float4 p1 = *(const float4*)(vr + 4);
      float4 p2 = *(const float4*)(vr + 16);
      float4 p3 = *(const float4*)(vr + 20);
      b0[0]=(__bf16)p0.x; b0[1]=(__bf16)p0.y; b0[2]=(__bf16)p0.z; b0[3]=(__bf16)p0.w;
      b0[4]=(__bf16)p1.x; b0[5]=(__bf16)p1.y; b0[6]=(__bf16)p1.z; b0[7]=(__bf16)p1.w;
      b0[8]=(__bf16)p2.x; b0[9]=(__bf16)p2.y; b0[10]=(__bf16)p2.z; b0[11]=(__bf16)p2.w;
      b0[12]=(__bf16)p3.x; b0[13]=(__bf16)p3.y; b0[14]=(__bf16)p3.z; b0[15]=(__bf16)p3.w;
      vr += (size_t)16 * NTOT;
      p0 = *(const float4*)(vr);
      p1 = *(const float4*)(vr + 4);
      p2 = *(const float4*)(vr + 16);
      p3 = *(const float4*)(vr + 20);
      b1[0]=(__bf16)p0.x; b1[1]=(__bf16)p0.y; b1[2]=(__bf16)p0.z; b1[3]=(__bf16)p0.w;
      b1[4]=(__bf16)p1.x; b1[5]=(__bf16)p1.y; b1[6]=(__bf16)p1.z; b1[7]=(__bf16)p1.w;
      b1[8]=(__bf16)p2.x; b1[9]=(__bf16)p2.y; b1[10]=(__bf16)p2.z; b1[11]=(__bf16)p2.w;
      b1[12]=(__bf16)p3.x; b1[13]=(__bf16)p3.y; b1[14]=(__bf16)p3.z; b1[15]=(__bf16)p3.w;
    }
    acc00 = wmma_bf16(a0, b0, acc00);  acc01 = wmma_bf16(a0, b1, acc01);
    acc10 = wmma_bf16(a1, b0, acc10);  acc11 = wmma_bf16(a1, b1, acc11);
    ks0   = wmma_bf16(a0, ones, ks0);  ks1   = wmma_bf16(a1, ones, ks1);
  }

  float* kvd = kvb + (size_t)bh * 1024;
#pragma unroll
  for (int r8 = 0; r8 < 8; ++r8) {
    int d0 = r8 + half * 8;
    atomicAdd(kvd + (size_t)d0 * 32 + r,             acc00[r8]);
    atomicAdd(kvd + (size_t)d0 * 32 + 16 + r,        acc01[r8]);
    atomicAdd(kvd + (size_t)(16 + d0) * 32 + r,      acc10[r8]);
    atomicAdd(kvd + (size_t)(16 + d0) * 32 + 16 + r, acc11[r8]);
  }
  if (r == 0) {   // ones-column result replicated across cols; take col 0
    float* ksd = ksum + (size_t)bh * 32;
#pragma unroll
    for (int r8 = 0; r8 < 8; ++r8) {
      atomicAdd(ksd + r8 + half * 8,      ks0[r8]);
      atomicAdd(ksd + 16 + r8 + half * 8, ks1[r8]);
    }
  }
}

// --------------------------------------------------- o^T = kv^T q' , / norm
__global__ void __launch_bounds__(256)
attn_out_kernel(const float* __restrict__ qsrc, long long qBStride, long long qHStride,
                const float* __restrict__ kvb, const float* __restrict__ ksum,
                const float* __restrict__ sw, int swIdx,
                float* __restrict__ dst, int Nq) {
  const int wave = (int)((blockIdx.x * blockDim.x + threadIdx.x) >> 5);
  const int lane = threadIdx.x & 31;
  const int half = lane >> 4;
  const int r    = lane & 15;
  const int ntiles = Nq >> 4;
  int bh = wave / ntiles;
  int nt = wave - bh * ntiles;
  int n0 = nt * 16;
  int b = bh >> 3, hh = bh & 7;
  const float* qb  = qsrc + (size_t)b * qBStride + (size_t)hh * qHStride;
  const float* kvd = kvb + (size_t)bh * 1024;

  // A = kv^T tiles: row e, K = d
  v16bf aT0, aT1;
#pragma unroll
  for (int i = 0; i < 8; ++i) {
    int d = half * 16 + i;
    aT0[i]     = (__bf16)kvd[(size_t)d * 32 + r];
    aT0[8 + i] = (__bf16)kvd[(size_t)(d + 8) * 32 + r];
    aT1[i]     = (__bf16)kvd[(size_t)d * 32 + 16 + r];
    aT1[8 + i] = (__bf16)kvd[(size_t)(d + 8) * 32 + 16 + r];
  }
  const float* ksd = ksum + (size_t)bh * 32 + half * 16;
  float4 k0v = *(const float4*)(ksd);
  float4 k1v = *(const float4*)(ksd + 4);
  float4 k2v = *(const float4*)(ksd + 8);
  float4 k3v = *(const float4*)(ksd + 12);
  float ksv[16] = {k0v.x, k0v.y, k0v.z, k0v.w, k1v.x, k1v.y, k1v.z, k1v.w,
                   k2v.x, k2v.y, k2v.z, k2v.w, k3v.x, k3v.y, k3v.z, k3v.w};
  float w = sw[swIdx];

  // B fragment from q (feature map applied on the fly) + norm partial dot
  v16bf bf;
  float normp = 0.f;
#pragma unroll
  for (int j = 0; j < 16; ++j) {
    float qv = qb[(size_t)(half * 16 + j) * Nq + n0 + r];
    float f  = featmap(qv);
    bf[j] = (__bf16)f;
    normp += f * ksv[j];
  }
  float norm = normp + __shfl_xor(normp, 16, 32);  // merge the two K halves
  norm = fmaxf(norm, 1e-6f);
  float s = w / norm;

  v8f z = {};
  v8f o0 = wmma_bf16(aT0, bf, z);
  v8f o1 = wmma_bf16(aT1, bf, z);

  float* dbase = dst + (size_t)(bh * 32 + half * 8) * Nq + n0 + r;
#pragma unroll
  for (int r8 = 0; r8 < 8; ++r8) {
    dbase[(size_t)r8 * Nq]        = o0[r8] * s;
    dbase[(size_t)(r8 + 16) * Nq] = o1[r8] * s;
  }
}

// --------------------------------------------------- adaptive avg pool (s x s)
__global__ void pool_kernel(const float* __restrict__ qkv, float* __restrict__ qp,
                            int s, int pH, int pW) {
  int idx = blockIdx.x * blockDim.x + threadIdx.x;
  int pN = pH * pW;
  int total = BH * DHEAD * pN;
  if (idx >= total) return;
  int c = idx / pN;               // c = b*256 + hh*32 + d
  int p = idx - c * pN;
  int py = p / pW, px = p - py * pW;
  int b = c >> 8;
  int chan = c & 255;
  const float* src = qkv + ((size_t)b * 3 * CC + chan) * NTOT;
  float acc = 0.f;
  for (int dy = 0; dy < s; ++dy) {
    const float* row = src + (size_t)(py * s + dy) * WW + px * s;
    for (int dx = 0; dx < s; ++dx) acc += row[dx];
  }
  qp[idx] = acc / (float)(s * s);
}

// --------------------------------------------------- bilinear upsample + add
__global__ void upsample_add_kernel(const float* __restrict__ ao,
                                    float* __restrict__ attn,
                                    int pH, int pW, float invs) {
  int idx = blockIdx.x * blockDim.x + threadIdx.x;
  int total = BH * DHEAD * NTOT;
  if (idx >= total) return;
  int c = idx >> 14;
  int p = idx & 16383;
  int y = p >> 7, x = p & 127;
  float fy = fmaxf((y + 0.5f) * invs - 0.5f, 0.f);
  float fx = fmaxf((x + 0.5f) * invs - 0.5f, 0.f);
  int y0 = (int)fy; if (y0 > pH - 1) y0 = pH - 1;
  int x0 = (int)fx; if (x0 > pW - 1) x0 = pW - 1;
  int y1 = min(y0 + 1, pH - 1);
  int x1 = min(x0 + 1, pW - 1);
  float wy = fy - y0, wx = fx - x0;
  const float* src = ao + (size_t)c * pH * pW;
  float v00 = src[y0 * pW + x0], v01 = src[y0 * pW + x1];
  float v10 = src[y1 * pW + x0], v11 = src[y1 * pW + x1];
  float v0 = v00 + (v01 - v00) * wx;
  float v1 = v10 + (v11 - v10) * wx;
  attn[idx] += v0 + (v1 - v0) * wy;
}

// ---------------------------------------------------------------------------
extern "C" void kernel_launch(void* const* d_in, const int* in_sizes, int n_in,
                              void* d_out, int out_size, void* d_ws, size_t ws_size,
                              hipStream_t stream) {
  (void)in_sizes; (void)n_in; (void)out_size; (void)ws_size;
  const float* x       = (const float*)d_in[0];
  const float* W_qkv   = (const float*)d_in[1];
  const float* W_proj  = (const float*)d_in[2];
  const float* scale_w = (const float*)d_in[3];
  float* out = (float*)d_out;
  float* ws  = (float*)d_ws;

  // workspace layout (floats)
  float* qkv  = ws;                 // 4*768*16384       = 50331648
  float* attn = ws + 50331648;      // 4*256*16384       = 16777216
  float* kvb  = ws + 67108864;      // 32*32*32          = 32768
  float* ksum = ws + 67141632;      // 32*32             = 1024
  float* sw   = ws + 67142656;      // 16 (padded)
  float* qp2  = ws + 67142672;      // 32*32*4096        = 4194304
  float* qp4  = ws + 71336976;      // 32*32*1024        = 1048576
  float* ao2  = ws + 72385552;      // 4194304
  float* ao4  = ws + 76579856;      // 1048576   (total 77628432 floats)

  softmax3_kernel<<<1, 32, 0, stream>>>(scale_w, sw);

  // qkv = W_qkv @ x : blocks = 4 * (768/64) * (16384/128) = 6144
  gemm_wmma_kernel<<<6144, 256, 0, stream>>>(W_qkv, x, qkv, 768, NTOT);

  hipMemsetAsync(kvb, 0, (32768 + 1024) * sizeof(float), stream);

  // kv / k_sum (shared across all scales): waves = 32*16 -> 64 blocks
  kv_reduce_kernel<<<64, 256, 0, stream>>>(qkv, kvb, ksum);

  // pooled q for scales 2 and 4
  {
    int t2 = BH * DHEAD * 64 * 64;
    pool_kernel<<<(t2 + 255) / 256, 256, 0, stream>>>(qkv, qp2, 2, 64, 64);
    int t4 = BH * DHEAD * 32 * 32;
    pool_kernel<<<(t4 + 255) / 256, 256, 0, stream>>>(qkv, qp4, 4, 32, 32);
  }

  // scale 1: writes sw[0]-weighted result directly into attn [B][C][N]
  attn_out_kernel<<<4096, 256, 0, stream>>>(qkv, (long long)768 * NTOT,
                                            (long long)32 * NTOT,
                                            kvb, ksum, sw, 0, attn, NTOT);
  // scale 2 / 4: sw-weighted into ao buffers (then bilinear-added)
  attn_out_kernel<<<1024, 256, 0, stream>>>(qp2, (long long)8 * 32 * 4096,
                                            (long long)32 * 4096,
                                            kvb, ksum, sw, 1, ao2, 4096);
  attn_out_kernel<<<256, 256, 0, stream>>>(qp4, (long long)8 * 32 * 1024,
                                           (long long)32 * 1024,
                                           kvb, ksum, sw, 2, ao4, 1024);

  int tU = BH * DHEAD * NTOT;  // 16,777,216
  upsample_add_kernel<<<tU / 256, 256, 0, stream>>>(ao2, attn, 64, 64, 0.5f);
  upsample_add_kernel<<<tU / 256, 256, 0, stream>>>(ao4, attn, 32, 32, 0.25f);

  // out = W_proj @ attn : blocks = 4 * (256/64) * (16384/128) = 2048
  gemm_wmma_kernel<<<2048, 256, 0, stream>>>(W_proj, attn, out, 256, NTOT);
}
